// SharedAtomModel_24180665876970
// MI455X (gfx1250) — matrix operand order repaired and checked by reading, not compiled, not run on hardware
//
#include <hip/hip_runtime.h>
#include <hip/hip_bf16.h>
#include <math.h>

typedef __attribute__((ext_vector_type(2)))  float  v2f;
typedef __attribute__((ext_vector_type(8)))  float  v8f;
typedef __attribute__((ext_vector_type(16))) __bf16 v16bf;

#define D_MODEL 512
#define DICT    16384
#define TOPK    32

// d_out flat layout (floats):
//  [0)        clean concept logits  1024x16
//  [16384)    clean relation logits 1024x6
//  [22528)    noisy concept logits  1024x16
//  [38912)    noisy relation logits 1024x6
//  [45056)    4 x (1024 x 16384) dense-sparse coeffs (enc order c, r, nc, nr)
#define OUT_SP_OFF   45056
#define SP_STRIDE    ((size_t)1024 * DICT)

// ---------------------------------------------------------------------------
// 1) pooled embedding mean: x[row, d] = mean_s emb[tok[row,s], d]
//    rows 0..1023 = clean, 1024..2047 = noisy
// ---------------------------------------------------------------------------
__global__ __launch_bounds__(256) void pool_kernel(const int* __restrict__ clean,
                                                   const int* __restrict__ noisy,
                                                   const float* __restrict__ emb,
                                                   float* __restrict__ x) {
  int row = blockIdx.x;
  const int* toks = (row < 1024) ? (clean + row * 32) : (noisy + (row - 1024) * 32);
  int t = threadIdx.x;
  for (int u = 0; u < 2; ++u) {
    int d = t + u * 256;
    float s = 0.f;
#pragma unroll 4
    for (int i = 0; i < 32; ++i) s += emb[(size_t)toks[i] * D_MODEL + d];
    x[(size_t)row * D_MODEL + d] = s * (1.0f / 32.0f);
  }
}

// ---------------------------------------------------------------------------
// 2) MLP GEMM in full fp32 via V_WMMA_F32_16X16X4_F32.
//    out = act(A @ W^T + bias), A:[2048,512], W:[512,512] row-major [out,in].
//    f32 WMMA layouts (wave32):
//      A 16x4:  lane(0-15)->M, VGPR v + 2*half -> K
//      B 4x16:  lane(0-15)->N, VGPR v + 2*half -> K
//      C 16x16: VGPR r -> M = r + 8*half, lane&15 -> N
// ---------------------------------------------------------------------------
__global__ __launch_bounds__(256) void mlp_wmma_kernel(const float* __restrict__ A,
                                                       const float* __restrict__ W,
                                                       const float* __restrict__ bias,
                                                       float* __restrict__ out,
                                                       int act) {
  int mt   = blockIdx.x;              // 0..127 (16-row tiles of 2048 rows)
  int wave = threadIdx.x >> 5;
  int lane = threadIdx.x & 31;
  int nt   = blockIdx.y * 8 + wave;   // 0..31 (16-col tiles of 512 outputs)
  int half = lane >> 4, l15 = lane & 15;

  const float* arow = A + (size_t)(mt * 16 + l15) * D_MODEL + 2 * half;
  const float* brow = W + (size_t)(nt * 16 + l15) * D_MODEL + 2 * half;

  v8f acc = {0.f, 0.f, 0.f, 0.f, 0.f, 0.f, 0.f, 0.f};
#pragma unroll 8
  for (int k = 0; k < D_MODEL; k += 4) {
    v2f av = *(const v2f*)(arow + k);
    v2f bv = *(const v2f*)(brow + k);
    acc = __builtin_amdgcn_wmma_f32_16x16x4_f32(false, av, false, bv,
                                                (short)0, acc, false, false);
  }
  int n = nt * 16 + l15;
  float bb = bias[n];
#pragma unroll
  for (int r = 0; r < 8; ++r) {
    int m = mt * 16 + half * 8 + r;
    float v = acc[r] + bb;
    if (act) {                        // silu via fast v_rcp_f32
      float e = __expf(-v);
      v = v * __builtin_amdgcn_rcpf(1.0f + e);
    }
    out[(size_t)m * D_MODEL + n] = v;
  }
}

// ---------------------------------------------------------------------------
// 3) Pack p (fp32 [2048,512]) into bf16 WMMA-A fragment layout.
//    pA[((mt*16+kc)*32 + lane)*16 + e]  where for 16x16x32 bf16 A:
//      M = lane&15,  K = kc*32 + (e&7) + 16*(e>>3) + 8*(lane>>4)
// ---------------------------------------------------------------------------
__global__ __launch_bounds__(256) void packA_kernel(const float* __restrict__ p,
                                                    __bf16* __restrict__ pA) {
  int idx  = blockIdx.x * 256 + threadIdx.x;    // 0 .. 2048*512-1
  int e    = idx & 15;
  int lane = (idx >> 4) & 31;
  int kc   = (idx >> 9) & 15;
  int mt   = idx >> 13;
  int k = kc * 32 + (e & 7) + 16 * (e >> 3) + 8 * (lane >> 4);
  int m = mt * 16 + (lane & 15);
  pA[idx] = (__bf16)p[(size_t)m * D_MODEL + k];
}

// ---------------------------------------------------------------------------
// 4) Pack combined weight (router + 0.3*delta) into bf16 WMMA-B layout.
//    pW[((nt*16+kc)*32 + lane)*16 + e]  where for 16x16x32 bf16 B (32x16 KxN):
//      K = kc*32 + (e&7) + 16*(e>>3) + 8*(lane>>4),  N = nt*16 + (lane&15)
// ---------------------------------------------------------------------------
__global__ __launch_bounds__(256) void packW_kernel(const float* __restrict__ router,
                                                    const float* __restrict__ delta,
                                                    __bf16* __restrict__ pW) {
  int idx  = blockIdx.x * 256 + threadIdx.x;    // 0 .. 16384*512-1
  int e    = idx & 15;
  int lane = (idx >> 4) & 31;
  int kc   = (idx >> 9) & 15;
  int nt   = idx >> 13;
  int k = kc * 32 + (e & 7) + 16 * (e >> 3) + 8 * (lane >> 4);
  int n = nt * 16 + (lane & 15);
  size_t si = (size_t)n * D_MODEL + k;
  pW[idx] = (__bf16)(router[si] + 0.3f * delta[si]);
}

// ---------------------------------------------------------------------------
// 5) DICT GEMM: coeffs = p @ Wcombined^T, bf16 WMMA, fp32 accumulate.
//    grid (4 mgroups, 16 cgroups, 4 encodes), 512 threads = 16 waves.
//    Wave w owns one 16-row tile (A fragments hoisted into 16 v16bf regs).
//    All 16 waves sweep the SAME 64 column tiles, so each 16KB B tile is
//    staged ONCE per block into LDS via GLOBAL_LOAD_ASYNC_TO_LDS_B128
//    (ASYNCcnt-tracked DMA, double buffered: tile i+1 streams in while
//    tile i feeds the WMMAs from ds_load_b128).
// ---------------------------------------------------------------------------
__global__ __launch_bounds__(512) void dict_gemm_kernel(const __bf16* __restrict__ pA,
                                                        const __bf16* __restrict__ Wc,
                                                        const __bf16* __restrict__ Wr,
                                                        float* __restrict__ dout) {
  __shared__ __attribute__((aligned(32))) __bf16 smem[2 * 8192];  // 2 x 16KB B tile
  int mg   = blockIdx.x;   // 0..3
  int cg   = blockIdx.y;   // 0..15
  int enc  = blockIdx.z;   // 0:c 1:r 2:nc 3:nr
  int wave = threadIdx.x >> 5;
  int lane = threadIdx.x & 31;
  int mt   = mg * 16 + wave;            // 0..63 row tile within this encode
  int gmt  = (enc >> 1) * 64 + mt;      // row tile within [2048,512] p matrix
  const __bf16* W = (enc & 1) ? Wr : Wc;

  // hoist all 16 A fragments for this wave's 16-row tile
  v16bf a[16];
  const __bf16* abase = pA + ((size_t)gmt * 512 + lane) * 16;
#pragma unroll
  for (int kc = 0; kc < 16; ++kc)
    a[kc] = *(const v16bf*)(abase + (size_t)kc * 512);

  float* orow = dout + OUT_SP_OFF + (size_t)enc * SP_STRIDE + (size_t)mt * 16 * DICT;
  int l15 = lane & 15, mh = (lane >> 4) * 8;

  // LDS byte offset of this thread's 32-byte slice (low 32 bits of flat addr
  // of a __shared__ pointer == allocation-relative LDS offset, ISA 10.2)
  unsigned lmine = (unsigned)(size_t)(&smem[0]) + threadIdx.x * 32u;
  int ct0 = cg * 64;

  // prologue: stage tile 0 into buffer 0 (each thread DMAs 32 bytes)
  {
    const char* g = (const char*)(W + (size_t)ct0 * 8192) + threadIdx.x * 32;
    asm volatile("global_load_async_to_lds_b128 %0, %1, off\n\t"
                 "global_load_async_to_lds_b128 %0, %1, off offset:16"
                 :: "v"(lmine), "v"(g) : "memory");
  }

  unsigned buf = 0;
  for (int i = 0; i < 64; ++i) {
    int ct = ct0 + i;
    if (i + 1 < 64) {
      // stream next tile into the other buffer, then wait only for the
      // OLDER tile (leave the 2 new async loads in flight)
      const char* g = (const char*)(W + (size_t)(ct + 1) * 8192) + threadIdx.x * 32;
      unsigned l = lmine + ((buf ^ 1u) ? 16384u : 0u);
      asm volatile("global_load_async_to_lds_b128 %0, %1, off\n\t"
                   "global_load_async_to_lds_b128 %0, %1, off offset:16"
                   :: "v"(l), "v"(g) : "memory");
      asm volatile("s_wait_asynccnt 0x2" ::: "memory");
    } else {
      asm volatile("s_wait_asynccnt 0x0" ::: "memory");
    }
    __syncthreads();   // all waves' portion of current tile landed in LDS

    const __bf16* bbuf = smem + buf * 8192;
    v8f acc = {0.f, 0.f, 0.f, 0.f, 0.f, 0.f, 0.f, 0.f};
#pragma unroll
    for (int kc = 0; kc < 16; ++kc) {
      v16bf b = *(const v16bf*)(bbuf + kc * 512 + lane * 16);  // ds_load_b128 x2
      acc = __builtin_amdgcn_wmma_f32_16x16x32_bf16(false, a[kc], false, b,
                                                    (short)0, acc, false, false);
    }
    int n = ct * 16 + l15;
#pragma unroll
    for (int r = 0; r < 8; ++r)
      orow[(size_t)(mh + r) * DICT + n] = acc[r];

    __syncthreads();   // everyone done reading this buffer before re-targeting it
    buf ^= 1u;
  }
}

// ---------------------------------------------------------------------------
// 6) Exact abs-top-32 per row via 4-pass radix select on |v| bit patterns,
//    ties -> lowest index (matches lax.top_k stability). Then zero the row
//    and scatter the 32 signed values; record (idx,val) lists for kernel 7.
// ---------------------------------------------------------------------------
__global__ __launch_bounds__(256) void topk_kernel(float* __restrict__ dout,
                                                   int* __restrict__ tki,
                                                   float* __restrict__ tkv) {
  int row = blockIdx.x;                 // 0..4095  (enc*1024 + r)
  float* base = dout + OUT_SP_OFF + (size_t)row * DICT;
  int t = threadIdx.x;

  float v[64];
#pragma unroll 8
  for (int j = 0; j < 64; ++j) v[j] = base[t + (j << 8)];

  __shared__ unsigned hist[256];
  __shared__ unsigned sprefix;
  __shared__ int sneed;
  if (t == 0) { sprefix = 0u; sneed = TOPK; }
  __syncthreads();

  for (int shift = 24; shift >= 0; shift -= 8) {
    hist[t] = 0u;
    __syncthreads();
    unsigned pfx = sprefix;
#pragma unroll 8
    for (int j = 0; j < 64; ++j) {
      unsigned key = __float_as_uint(v[j]) & 0x7fffffffu;
      if (shift == 24 || (key >> (shift + 8)) == (pfx >> (shift + 8)))
        atomicAdd(&hist[(key >> shift) & 255u], 1u);
    }
    __syncthreads();
    if (t == 0) {
      int need = sneed;
      for (int b = 255; b >= 0; --b) {
        int c = (int)hist[b];
        if (c >= need) { sprefix = pfx | ((unsigned)b << shift); break; }
        need -= c;
      }
      sneed = need;
    }
    __syncthreads();
  }
  unsigned T = sprefix;
  int need = sneed;

  __shared__ int nsel, ntie;
  __shared__ int   sidx[TOPK];
  __shared__ float sval[TOPK];
  __shared__ int   tidx[64];
  if (t == 0) { nsel = 0; ntie = 0; }
  if (t < TOPK) { sidx[t] = 0; sval[t] = 0.f; }
  __syncthreads();

#pragma unroll 8
  for (int j = 0; j < 64; ++j) {
    unsigned key = __float_as_uint(v[j]) & 0x7fffffffu;
    int col = t + (j << 8);
    if (key > T) {
      int s = atomicAdd(&nsel, 1);
      if (s < TOPK) { sidx[s] = col; sval[s] = v[j]; }
    } else if (key == T) {
      int s = atomicAdd(&ntie, 1);
      if (s < 64) tidx[s] = col;
    }
  }
  __syncthreads();

  if (t == 0) {
    int nt_ = ntie < 64 ? ntie : 64;
    for (int i = 1; i < nt_; ++i) {       // sort tie indices ascending
      int x = tidx[i], j = i - 1;
      while (j >= 0 && tidx[j] > x) { tidx[j + 1] = tidx[j]; --j; }
      tidx[j + 1] = x;
    }
    int s0 = nsel < TOPK ? nsel : TOPK;
    for (int q = 0; q < need && q < nt_ && s0 + q < TOPK; ++q) {
      int col = tidx[q];
      sidx[s0 + q] = col;
      sval[s0 + q] = base[col];
    }
  }
  __syncthreads();

  if (t < TOPK) {
    tki[(size_t)row * TOPK + t] = sidx[t];
    tkv[(size_t)row * TOPK + t] = sval[t];
  }

  // zero the dense row, then scatter the selected signed values
  float4 z = {0.f, 0.f, 0.f, 0.f};
  float* zb = base + t * 64;
#pragma unroll
  for (int j = 0; j < 16; ++j) ((float4*)zb)[j] = z;
  __syncthreads();
  if (t < TOPK) base[sidx[t]] = sval[t];
}

// ---------------------------------------------------------------------------
// 7) fused = 0.2*p + 0.9*(sum_c val*atom) + 0.9*(sum_r val*atom); then heads.
// ---------------------------------------------------------------------------
__global__ __launch_bounds__(256) void fuse_head_kernel(const float* __restrict__ p,
                                                        const float* __restrict__ atoms,
                                                        const int* __restrict__ tki,
                                                        const float* __restrict__ tkv,
                                                        const float* __restrict__ chw,
                                                        const float* __restrict__ chb,
                                                        const float* __restrict__ rhw,
                                                        const float* __restrict__ rhb,
                                                        float* __restrict__ dout) {
  int b = blockIdx.x;                 // 0..2047
  int noisy = b >> 10, r = b & 1023;
  int t = threadIdx.x;

  __shared__ int   ci[TOPK]; __shared__ float cv[TOPK];
  __shared__ int   ri[TOPK]; __shared__ float rv[TOPK];
  __shared__ float fused[D_MODEL];

  if (t < TOPK) {
    size_t oc  = ((size_t)(noisy * 2) * 1024 + r) * TOPK + t;
    size_t orr = ((size_t)(noisy * 2 + 1) * 1024 + r) * TOPK + t;
    ci[t] = tki[oc];  cv[t] = tkv[oc];
    ri[t] = tki[orr]; rv[t] = tkv[orr];
  }
  __syncthreads();

  for (int u = 0; u < 2; ++u) {
    int d = t + u * 256;
    float acc = 0.2f * p[(size_t)b * D_MODEL + d];
#pragma unroll 4
    for (int j = 0; j < TOPK; ++j) acc += 0.9f * cv[j] * atoms[(size_t)ci[j] * D_MODEL + d];
#pragma unroll 4
    for (int j = 0; j < TOPK; ++j) acc += 0.9f * rv[j] * atoms[(size_t)ri[j] * D_MODEL + d];
    fused[d] = acc;
  }
  __syncthreads();

  int wave = t >> 5, lane = t & 31;
  for (int o = wave; o < 22; o += 8) {
    const float* hw; float bb; float* op;
    if (o < 16) {
      hw = chw + (size_t)o * D_MODEL; bb = chb[o];
      op = dout + (noisy ? (size_t)22528 : (size_t)0) + (size_t)r * 16 + o;
    } else {
      int o2 = o - 16;
      hw = rhw + (size_t)o2 * D_MODEL; bb = rhb[o2];
      op = dout + (noisy ? (size_t)38912 : (size_t)16384) + (size_t)r * 6 + o2;
    }
    float s = 0.f;
    for (int d = lane; d < D_MODEL; d += 32) s += fused[d] * hw[d];
    for (int off = 16; off > 0; off >>= 1) s += __shfl_xor(s, off, 32);
    if (lane == 0) *op = s + bb;
  }
}

// ---------------------------------------------------------------------------
// Launch. Workspace layout (needs ~47 MB of d_ws):
//   [0,4M) x   [4M,8M) h   [8M,12M) p   [12M,14M) pA(bf16)
//   [14M,30M) Wc(bf16)   [30M,46M) Wr(bf16)
//   [46M, +512K) tk_idx  [+512K, +1M) tk_val
// ---------------------------------------------------------------------------
extern "C" void kernel_launch(void* const* d_in, const int* in_sizes, int n_in,
                              void* d_out, int out_size, void* d_ws, size_t ws_size,
                              hipStream_t stream) {
  const int*   clean  = (const int*)d_in[0];
  const int*   noisy  = (const int*)d_in[1];
  const float* emb    = (const float*)d_in[2];
  const float* w1     = (const float*)d_in[3];
  const float* b1     = (const float*)d_in[4];
  const float* w2     = (const float*)d_in[5];
  const float* b2     = (const float*)d_in[6];
  const float* atoms  = (const float*)d_in[7];
  const float* router = (const float*)d_in[8];
  const float* cdel   = (const float*)d_in[9];
  const float* rdel   = (const float*)d_in[10];
  const float* chw    = (const float*)d_in[11];
  const float* chb    = (const float*)d_in[12];
  const float* rhw    = (const float*)d_in[13];
  const float* rhb    = (const float*)d_in[14];
  float* out = (float*)d_out;

  char* ws = (char*)d_ws;
  float*  x   = (float*)(ws + ((size_t)0 << 20));
  float*  h   = (float*)(ws + ((size_t)4 << 20));
  float*  p   = (float*)(ws + ((size_t)8 << 20));
  __bf16* pA  = (__bf16*)(ws + ((size_t)12 << 20));
  __bf16* Wc  = (__bf16*)(ws + ((size_t)14 << 20));
  __bf16* Wr  = (__bf16*)(ws + ((size_t)30 << 20));
  int*    tki = (int*)   (ws + ((size_t)46 << 20));
  float*  tkv = (float*) (ws + ((size_t)46 << 20) + ((size_t)512 << 10));

  pool_kernel<<<2048, 256, 0, stream>>>(clean, noisy, emb, x);
  mlp_wmma_kernel<<<dim3(128, 4), 256, 0, stream>>>(x, w1, b1, h, 1);
  mlp_wmma_kernel<<<dim3(128, 4), 256, 0, stream>>>(h, w2, b2, p, 0);
  packA_kernel<<<4096, 256, 0, stream>>>(p, pA);
  packW_kernel<<<32768, 256, 0, stream>>>(router, cdel, Wc);
  packW_kernel<<<32768, 256, 0, stream>>>(router, rdel, Wr);
  dict_gemm_kernel<<<dim3(4, 16, 4), 512, 0, stream>>>(pA, Wc, Wr, out);
  topk_kernel<<<4096, 256, 0, stream>>>(out, tki, tkv);
  fuse_head_kernel<<<2048, 256, 0, stream>>>(p, atoms, tki, tkv, chw, chb, rhw, rhb, out);
}
